// OnetoManyGRU_49821620634112
// MI455X (gfx1250) — matrix operand order repaired and checked by reading, not compiled, not run on hardware
//
#include <hip/hip_runtime.h>
#include <hip/hip_bf16.h>

// ---------------------------------------------------------------------------
// OneToMany GRU on gfx1250 (MI455X), f16 WMMA + fp32 accumulation.
//   B=1024, E=1024, K=64.
// Per step: G = [x_k ; h] @ [W_ih W_hh]^T; r,z = sigmoid; n = tanh(i_n + r*h_n);
//           h' = (1-z)*n + z*h;  y[:, :, k] = h'.
// v2: software-pipelined (double-buffered) fragment loads, 2 M-tiles per wave
//     (B-fragment reuse), branch-free templated step kernels.
// ---------------------------------------------------------------------------

typedef _Float16 v16h __attribute__((ext_vector_type(16)));
typedef _Float16 v8h  __attribute__((ext_vector_type(8)));
typedef float    v8f  __attribute__((ext_vector_type(8)));

constexpr int Bsz = 1024;   // batch
constexpr int E   = 1024;   // hidden
constexpr int K   = 64;     // steps
constexpr int NCT = E / 16; // 64 col-tiles per gate
constexpr int NKC = E / 32; // 32 k-chunks over E

#define WMMA_F16(a, b, c) \
    __builtin_amdgcn_wmma_f32_16x16x32_f16(false, (a), false, (b), (short)0, (c), false, false)

// ---------------------------------------------------------------------------
// Pack W (3E x E, row-major fp32) into f16 WMMA B-fragments.
// Fragment (g, ct, kc): element (k = kc*32 + kk, n) taken from
//   W[(g*E + ct*16 + n) * E + kc*32 + kk]
// lane = n + 16*((kk/8)&1), half index = (kk%8) + 8*(kk>=16) (table order).
// ---------------------------------------------------------------------------
__global__ void gru_pack_w(const float* __restrict__ W, _Float16* __restrict__ dst) {
    int t    = blockIdx.x * blockDim.x + threadIdx.x;   // 3*NCT*NKC*32 threads
    int lane = t & 31;
    int fi   = t >> 5;                                  // (g*NCT + ct)*NKC + kc
    int kc   = fi & (NKC - 1);
    int ct   = (fi >> 5) & (NCT - 1);
    int g    = fi >> 11;
    int n    = lane & 15;
    int kh   = (lane >> 4) * 8;                         // 0 or 8
    const float* src = W + (size_t)(g * E + ct * 16 + n) * E + kc * 32 + kh;
    v16h out;
#pragma unroll
    for (int i = 0; i < 16; ++i) out[i] = (_Float16)src[i < 8 ? i : i + 8];
    *(v16h*)(dst + (size_t)t * 16) = out;
}

// x (B, E, K) fp32 -> XT (K-1, B*E) f16 row-major: XT[s][b*E+e] = x[b][e][s]
__global__ void gru_pack_x(const float* __restrict__ x, _Float16* __restrict__ xt) {
    int t = blockIdx.x * blockDim.x + threadIdx.x;      // B*E threads
    const float* src = x + (size_t)t * K;
    const size_t BE = (size_t)Bsz * E;
#pragma unroll 4
    for (int so = 0; so < K - 1; so += 4) {
        float4 v = *(const float4*)(src + so);
        if (so + 0 < K - 1) xt[(size_t)(so + 0) * BE + t] = (_Float16)v.x;
        if (so + 1 < K - 1) xt[(size_t)(so + 1) * BE + t] = (_Float16)v.y;
        if (so + 2 < K - 1) xt[(size_t)(so + 2) * BE + t] = (_Float16)v.z;
        if (so + 3 < K - 1) xt[(size_t)(so + 3) * BE + t] = (_Float16)v.w;
    }
}

__global__ void gru_init_h(const float* __restrict__ c, _Float16* __restrict__ h0) {
    int t = blockIdx.x * blockDim.x + threadIdx.x;      // B*E threads
    h0[t] = (_Float16)c[t];
}

// A-fragment load: lane holds row (lane&15), K-halves {sel..sel+7, sel+16..sel+23}
__device__ __forceinline__ v16h load_a(const _Float16* p) {
    v8h lo = *(const v8h*)(p);
    v8h hi = *(const v8h*)(p + 16);
    return __builtin_shufflevector(lo, hi, 0,1,2,3,4,5,6,7,8,9,10,11,12,13,14,15);
}

// ---------------------------------------------------------------------------
// One GRU step. Block = 256 threads = 8 waves (wave32).
//   blockIdx.x*8 + wave = N-tile (e/16) in [0,64)
//   blockIdx.y          = M-pair (batch/32): each wave owns 2 M-tiles.
// Double-buffered fragment pipeline: loads for chunk kc+1 issue before the
// WMMAs of chunk kc so v_wmma overlaps global_load_b128 latency.
// ---------------------------------------------------------------------------
template <bool HAS_X>
__device__ __forceinline__ void
gru_step_impl(const _Float16* __restrict__ xk, const _Float16* __restrict__ hcur,
              _Float16* __restrict__ hnext, const _Float16* __restrict__ wih,
              const _Float16* __restrict__ whh, const float* __restrict__ bih,
              const float* __restrict__ bhh, float* __restrict__ y, int k) {
    const int lane = threadIdx.x & 31;
    const int wave = threadIdx.x >> 5;
    const int nt   = blockIdx.x * 8 + wave;   // N-tile
    const int m0   = blockIdx.y * 32;         // first batch row of the M-pair
    const int row  = lane & 15;
    const int hi16 = lane >> 4;
    const int asel = hi16 * 8;

    const size_t GST = (size_t)NCT * NKC * 512;   // halves per gate in packed W
    const size_t fb0 = (size_t)nt * NKC * 512 + (size_t)lane * 16;

    const _Float16* ah0p = hcur + (size_t)(m0 + row) * E + asel;
    const _Float16* ah1p = ah0p + (size_t)16 * E;
    const _Float16* ax0p = xk   + (size_t)(m0 + row) * E + asel;
    const _Float16* ax1p = ax0p + (size_t)16 * E;

    v8f accr0 = {}, accz0 = {}, accn0 = {}, acch0 = {};
    v8f accr1 = {}, accz1 = {}, accn1 = {}, acch1 = {};

    // ---- prologue: fragments for kc = 0 ----
    v16h bhr = *(const v16h*)(whh + fb0);
    v16h bhz = *(const v16h*)(whh + fb0 + GST);
    v16h bhn = *(const v16h*)(whh + fb0 + 2 * GST);
    v16h ah0 = load_a(ah0p);
    v16h ah1 = load_a(ah1p);
    v16h bir, biz, bin, ax0, ax1;
    if constexpr (HAS_X) {
        bir = *(const v16h*)(wih + fb0);
        biz = *(const v16h*)(wih + fb0 + GST);
        bin = *(const v16h*)(wih + fb0 + 2 * GST);
        ax0 = load_a(ax0p);
        ax1 = load_a(ax1p);
    }

    for (int kc = 0; kc < NKC - 1; ++kc) {
        // ---- issue next chunk's loads (overlap with this chunk's WMMAs) ----
        const int    kb = (kc + 1) * 32;
        const size_t fb = fb0 + (size_t)(kc + 1) * 512;
        v16h n_bhr = *(const v16h*)(whh + fb);
        v16h n_bhz = *(const v16h*)(whh + fb + GST);
        v16h n_bhn = *(const v16h*)(whh + fb + 2 * GST);
        v16h n_ah0 = load_a(ah0p + kb);
        v16h n_ah1 = load_a(ah1p + kb);
        v16h n_bir, n_biz, n_bin, n_ax0, n_ax1;
        if constexpr (HAS_X) {
            n_bir = *(const v16h*)(wih + fb);
            n_biz = *(const v16h*)(wih + fb + GST);
            n_bin = *(const v16h*)(wih + fb + 2 * GST);
            n_ax0 = load_a(ax0p + kb);
            n_ax1 = load_a(ax1p + kb);
        }

        // ---- WMMAs for current chunk (RAW chains separated by >=5 ops) ----
        accr0 = WMMA_F16(ah0, bhr, accr0);
        accz0 = WMMA_F16(ah0, bhz, accz0);
        acch0 = WMMA_F16(ah0, bhn, acch0);
        accr1 = WMMA_F16(ah1, bhr, accr1);
        accz1 = WMMA_F16(ah1, bhz, accz1);
        acch1 = WMMA_F16(ah1, bhn, acch1);
        if constexpr (HAS_X) {
            accr0 = WMMA_F16(ax0, bir, accr0);
            accz0 = WMMA_F16(ax0, biz, accz0);
            accn0 = WMMA_F16(ax0, bin, accn0);
            accr1 = WMMA_F16(ax1, bir, accr1);
            accz1 = WMMA_F16(ax1, biz, accz1);
            accn1 = WMMA_F16(ax1, bin, accn1);
        }

        // ---- rotate buffers ----
        bhr = n_bhr; bhz = n_bhz; bhn = n_bhn; ah0 = n_ah0; ah1 = n_ah1;
        if constexpr (HAS_X) {
            bir = n_bir; biz = n_biz; bin = n_bin; ax0 = n_ax0; ax1 = n_ax1;
        }
    }

    // ---- epilogue chunk ----
    accr0 = WMMA_F16(ah0, bhr, accr0);
    accz0 = WMMA_F16(ah0, bhz, accz0);
    acch0 = WMMA_F16(ah0, bhn, acch0);
    accr1 = WMMA_F16(ah1, bhr, accr1);
    accz1 = WMMA_F16(ah1, bhz, accz1);
    acch1 = WMMA_F16(ah1, bhn, acch1);
    if constexpr (HAS_X) {
        accr0 = WMMA_F16(ax0, bir, accr0);
        accz0 = WMMA_F16(ax0, biz, accz0);
        accn0 = WMMA_F16(ax0, bin, accn0);
        accr1 = WMMA_F16(ax1, bir, accr1);
        accz1 = WMMA_F16(ax1, biz, accz1);
        accn1 = WMMA_F16(ax1, bin, accn1);
    }

    // ---- gating epilogue. C layout: VGPR v, lanes 0-15 -> M=v, 16-31 -> M=v+8.
    const int n = lane & 15;
    const int e = nt * 16 + n;
    const float br   = bih[e]         + bhh[e];
    const float bz   = bih[E + e]     + bhh[E + e];
    const float bin_ = bih[2 * E + e];
    const float bhn_ = bhh[2 * E + e];

    auto finish = [&](const v8f& ar, const v8f& az, const v8f& an, const v8f& ahn,
                      int mbase) {
#pragma unroll
        for (int v = 0; v < 8; ++v) {
            const int m = mbase + v + hi16 * 8;
            const float r  = 1.0f / (1.0f + __expf(-(ar[v] + br)));
            const float z  = 1.0f / (1.0f + __expf(-(az[v] + bz)));
            const float nn = tanhf(an[v] + bin_ + r * (ahn[v] + bhn_));
            const float hp = (float)hcur[(size_t)m * E + e];
            const float hv = (1.0f - z) * nn + z * hp;
            hnext[(size_t)m * E + e] = (_Float16)hv;
            y[((size_t)m * E + e) * K + k] = hv;
        }
    };
    finish(accr0, accz0, accn0, acch0, m0);
    finish(accr1, accz1, accn1, acch1, m0 + 16);
}

__global__ void __launch_bounds__(256)
gru_step_first(const _Float16* __restrict__ hcur, _Float16* __restrict__ hnext,
               const _Float16* __restrict__ whh, const float* __restrict__ bih,
               const float* __restrict__ bhh, float* __restrict__ y) {
    gru_step_impl<false>(nullptr, hcur, hnext, nullptr, whh, bih, bhh, y, 0);
}

__global__ void __launch_bounds__(256)
gru_step_body(const _Float16* __restrict__ xk, const _Float16* __restrict__ hcur,
              _Float16* __restrict__ hnext, const _Float16* __restrict__ wih,
              const _Float16* __restrict__ whh, const float* __restrict__ bih,
              const float* __restrict__ bhh, float* __restrict__ y, int k) {
    gru_step_impl<true>(xk, hcur, hnext, wih, whh, bih, bhh, y, k);
}

// ---------------------------------------------------------------------------
extern "C" void kernel_launch(void* const* d_in, const int* in_sizes, int n_in,
                              void* d_out, int out_size, void* d_ws, size_t ws_size,
                              hipStream_t stream) {
    const float* c    = (const float*)d_in[0];   // (B,E)
    const float* x    = (const float*)d_in[1];   // (B,E,K)
    const float* W_ih = (const float*)d_in[2];   // (3E,E)
    const float* W_hh = (const float*)d_in[3];   // (3E,E)
    const float* b_ih = (const float*)d_in[4];   // (3E)
    const float* b_hh = (const float*)d_in[5];   // (3E)
    float* y = (float*)d_out;                    // (B,E,K)

    // Workspace carve-up (~149 MB total)
    char* ws = (char*)d_ws;
    _Float16* wihp = (_Float16*)ws;                              // 3E*E halves = 6 MB
    _Float16* whhp = (_Float16*)(ws + 6291456);                  // 6 MB
    _Float16* h0   = (_Float16*)(ws + 12582912);                 // 2 MB
    _Float16* h1   = (_Float16*)(ws + 14680064);                 // 2 MB
    _Float16* xt   = (_Float16*)(ws + 16777216);                 // (K-1)*B*E = 126 MB

    // One-time prep (stream-ordered, graph-capture safe)
    const int packThreads = 3 * NCT * NKC * 32;                  // 196608
    gru_pack_w<<<packThreads / 256, 256, 0, stream>>>(W_ih, wihp);
    gru_pack_w<<<packThreads / 256, 256, 0, stream>>>(W_hh, whhp);
    gru_pack_x<<<(Bsz * E) / 256, 256, 0, stream>>>(x, xt);
    gru_init_h<<<(Bsz * E) / 256, 256, 0, stream>>>(c, h0);

    // Sequential recurrence: 64 step kernels, h ping-pong
    dim3 grid(NCT / 8, Bsz / 32);   // (8, 32)
    gru_step_first<<<grid, 256, 0, stream>>>(h0, h1, whhp, b_ih, b_hh, y);
    _Float16* hc = h1;
    _Float16* hn = h0;
    for (int k = 1; k < K; ++k) {
        const _Float16* xk = xt + (size_t)(k - 1) * Bsz * E;
        gru_step_body<<<grid, 256, 0, stream>>>(xk, hc, hn, wihp, whhp, b_ih, b_hh, y, k);
        _Float16* t = hc; hc = hn; hn = t;
    }
}
